// Attention_62345745269328
// MI455X (gfx1250) — matrix operand order repaired and checked by reading, not compiled, not run on hardware
//
#include <hip/hip_runtime.h>

#define HEADS 4
#define DHEAD 32
#define CDIM  256
#define HID   128
#define SEQ   4096
#define BATCH 4
#define QSCALE 0.17677669529663687f  // 32^-0.5
#define NCHUNK (SEQ / 32)

typedef _Float16 v16h __attribute__((ext_vector_type(16)));
typedef _Float16 h8   __attribute__((ext_vector_type(8)));
typedef float    v8f  __attribute__((ext_vector_type(8)));

__device__ __forceinline__ v8f wmma16(v16h a, v16h b, v8f c) {
  // D = A(16x32 f16) * B(32x16 f16) + C(16x16 f32)
  return __builtin_amdgcn_wmma_f32_16x16x32_f16(false, a, false, b, (short)0, c,
                                                false, false);
}

__device__ __forceinline__ v16h cat8(h8 a, h8 b) {
  return __builtin_shufflevector(a, b, 0, 1, 2, 3, 4, 5, 6, 7, 8, 9, 10, 11, 12,
                                 13, 14, 15);
}

// Build an A-fragment: `row` points at 16 contiguous K halves (chunk1 at +8).
__device__ __forceinline__ v16h loadA(const _Float16* row) {
  h8 a0 = *(const h8*)(row);
  h8 a1 = *(const h8*)(row + 8);
  return cat8(a0, a1);
}

// ---------------------------------------------------------------------------
// Kernel 1: convert weights to f16 (fold softmax scale into W_q rows 0..127)
// ---------------------------------------------------------------------------
__global__ void conv_weights(const float* __restrict__ wqkv,
                             const float* __restrict__ wout,
                             _Float16* __restrict__ wq,
                             _Float16* __restrict__ wo) {
  int i = blockIdx.x * 256 + threadIdx.x;
  if (i < 384 * 256) {
    float v = wqkv[i];
    if (i < 128 * 256) v *= QSCALE;  // q rows pre-scaled
    wq[i] = (_Float16)v;
  }
  if (i < 256 * 128) wo[i] = (_Float16)wout[i];
}

// ---------------------------------------------------------------------------
// Kernel 2: x[b,c,s] f32 -> xt[b,s,c] f16 (LDS tile transpose, coalesced both ways)
// ---------------------------------------------------------------------------
__global__ void xpose(const float* __restrict__ x, _Float16* __restrict__ xt) {
  __shared__ _Float16 t[32][34];
  const int b = blockIdx.z, c0 = blockIdx.y * 32, s0 = blockIdx.x * 32;
  const int tx = threadIdx.x, ty = threadIdx.y;
  t[ty][tx] = (_Float16)x[((size_t)b * CDIM + c0 + ty) * SEQ + s0 + tx];
  __syncthreads();
  xt[((size_t)b * SEQ + s0 + ty) * CDIM + c0 + tx] = t[tx][ty];
}

// ---------------------------------------------------------------------------
// Kernel 3: QKV projection GEMM (M=384, K=256, N=4096 per batch), WMMA f16.
// 8 waves of a block share the same (b, s-group) so the big B operand (xt)
// is identical across waves (WGP$/L2 hits); only the small A (weights) differs.
// Writes q[b,h,s,d], k[b,h,s,d], vT[b,h,d,s] in f16.
// ---------------------------------------------------------------------------
__global__ void __launch_bounds__(256) qkv_gemm(const _Float16* __restrict__ xt,
                                                const _Float16* __restrict__ wq,
                                                _Float16* __restrict__ q,
                                                _Float16* __restrict__ k,
                                                _Float16* __restrict__ vt) {
  const int lane = threadIdx.x & 31;
  const int lo = lane & 15, hi = lane >> 4;
  const int warp = threadIdx.x >> 5;
  const int mg = blockIdx.x % 3;               // 3 m-groups of 8 tiles
  const int ng = (blockIdx.x / 3) & 63;        // 64 s-groups of width 64
  const int b  = blockIdx.x / (3 * 64);
  const int mt = mg * 8 + warp;                // 24 tiles over M=384
  const int o0 = mt * 16;
  const int s0 = ng * 64;

  v8f acc[4] = {};
  for (int ks = 0; ks < 8; ++ks) {  // K = 256 in 8 steps of 32
    const int c0 = ks * 32;
    v16h a = loadA(wq + (size_t)(o0 + lo) * CDIM + c0 + 16 * hi);
#pragma unroll
    for (int sub = 0; sub < 4; ++sub) {
      const int s = s0 + sub * 16 + lo;
      v16h bv = *(const v16h*)(xt + ((size_t)b * SEQ + s) * CDIM + c0 + 16 * hi);
      acc[sub] = wmma16(a, bv, acc[sub]);
    }
  }

  const int grp    = o0 >> 7;        // 0=q, 1=k, 2=v
  const int within = o0 & 127;
  const int head   = within >> 5;
  const int dbase  = (within & 16) + 8 * hi;  // d for accumulator row r is dbase+r
#pragma unroll
  for (int sub = 0; sub < 4; ++sub) {
    const int s = s0 + sub * 16 + lo;
    if (grp < 2) {
      h8 st = {};
#pragma unroll
      for (int r = 0; r < 8; ++r) st[r] = (_Float16)acc[sub][r];
      _Float16* dst = (grp == 0 ? q : k) +
                      (((size_t)b * HEADS + head) * SEQ + s) * DHEAD + dbase;
      *(h8*)dst = st;
    } else {  // V stored transposed [b,h,d,s]
#pragma unroll
      for (int r = 0; r < 8; ++r)
        vt[(((size_t)b * HEADS + head) * DHEAD + dbase + r) * SEQ + s] =
            (_Float16)acc[sub][r];
    }
  }
}

// ---------------------------------------------------------------------------
// Kernel 4: flash attention. Block = 8 waves, all on the same (b, head); each
// wave owns one 16-query tile. Each 32-key K/V chunk (4KB) is staged into LDS
// ONCE per block (double-buffered, one barrier/chunk) and shared by all 8
// waves: 8x less global K/V traffic. Q fragment lives in registers for the
// whole key loop (d=32 == one WMMA K-step). Online softmax; P goes
// C-layout -> A-layout through a 1KB/wave LDS tile. global_prefetch_b8 pulls
// chunk c+2 while chunk c computes.
// ---------------------------------------------------------------------------
__global__ void __launch_bounds__(256) flash_attn(const _Float16* __restrict__ qb,
                                                  const _Float16* __restrict__ kb,
                                                  const _Float16* __restrict__ vt,
                                                  _Float16* __restrict__ att) {
  // [buf][ K tile: key*32+d (1024) | V tile: d*32+key (1024) ]
  __shared__ _Float16 kv[2][2048];   // 8 KB
  __shared__ _Float16 Pb[8 * 512];   // 8 KB, 16x32 f16 per wave
  const int t    = threadIdx.x;
  const int lane = t & 31;
  const int warp = t >> 5;
  const int lo = lane & 15, hi = lane >> 4;
  const int bh = blockIdx.x >> 5;          // 16 (b,h) pairs
  const int qt = (blockIdx.x & 31) * 8 + warp;
  _Float16* P = &Pb[warp * 512];

  const size_t qbase = ((size_t)bh * SEQ + (size_t)qt * 16) * DHEAD;
  const size_t kbase = (size_t)bh * SEQ * DHEAD;
  const size_t vbase = (size_t)bh * DHEAD * SEQ;

  // cooperative 4KB chunk stage: 128 threads for K (contiguous), 128 for V rows
  auto stage = [&](int c, int buf) {
    const int j0 = c * 32;
    if (t < 128) {
      h8 tmp = *(const h8*)(kb + kbase + (size_t)j0 * DHEAD + t * 8);
      *(h8*)&kv[buf][t * 8] = tmp;
    } else {
      const int u = t - 128;
      const int d = u >> 2, co = (u & 3) * 8;
      h8 tmp = *(const h8*)(vt + vbase + (size_t)d * SEQ + j0 + co);
      *(h8*)&kv[buf][1024 + d * 32 + co] = tmp;
    }
  };

  const v16h aq = loadA(qb + qbase + (size_t)lo * DHEAD + 16 * hi);

  v8f o0 = {}, o1 = {};
  float m[8], l[8];
#pragma unroll
  for (int r = 0; r < 8; ++r) { m[r] = -1e30f; l[r] = 0.f; }

  stage(0, 0);
  __syncthreads();

  for (int c = 0; c < NCHUNK; ++c) {
    const int buf = c & 1;
    if (c + 1 < NCHUNK) stage(c + 1, buf ^ 1);
    if (c + 2 < NCHUNK && (t & 7) == 0) {  // hint chunk c+2 toward L2
      const int j2 = (c + 2) * 32;
      if (t < 128)
        __builtin_prefetch(kb + kbase + (size_t)j2 * DHEAD + t * 8, 0, 1);
      else
        __builtin_prefetch(vt + vbase + (size_t)((t - 128) >> 2) * SEQ + j2, 0, 1);
    }

    // B fragments from LDS (32B each per lane, bank-friendly)
    v16h k0 = *(const v16h*)&kv[buf][lo * 32 + 16 * hi];
    v16h k1 = *(const v16h*)&kv[buf][(16 + lo) * 32 + 16 * hi];
    v8f z = {};
    v8f s0 = wmma16(aq, k0, z);
    v8f s1 = wmma16(aq, k1, z);

    // row-wise max over this 32-key chunk (rows 8*hi..8*hi+7 across 16 lanes)
    float mloc[8];
#pragma unroll
    for (int r = 0; r < 8; ++r) mloc[r] = fmaxf(s0[r], s1[r]);
#pragma unroll
    for (int mask = 1; mask < 16; mask <<= 1)
#pragma unroll
      for (int r = 0; r < 8; ++r)
        mloc[r] = fmaxf(mloc[r], __shfl_xor(mloc[r], mask, 32));

    float scl[8];
#pragma unroll
    for (int r = 0; r < 8; ++r) {
      float mn = fmaxf(m[r], mloc[r]);
      scl[r] = __expf(m[r] - mn);
      m[r] = mn;
    }
    // P = exp(S - m): store C-layout into per-wave LDS [16 x 32]
#pragma unroll
    for (int r = 0; r < 8; ++r) {
      float p0 = __expf(s0[r] - m[r]);
      float p1 = __expf(s1[r] - m[r]);
      l[r] = l[r] * scl[r] + p0 + p1;
      const int row = 8 * hi + r;
      P[row * 32 + lo]      = (_Float16)p0;
      P[row * 32 + 16 + lo] = (_Float16)p1;
    }
    asm volatile("s_wait_dscnt 0" ::: "memory");
    v16h pa = loadA(P + lo * 32 + 16 * hi);  // read back in A-layout

#pragma unroll
    for (int r = 0; r < 8; ++r) { o0[r] *= scl[r]; o1[r] *= scl[r]; }
    v16h v0 = *(const v16h*)&kv[buf][1024 + lo * 32 + 16 * hi];
    v16h v1 = *(const v16h*)&kv[buf][1024 + (16 + lo) * 32 + 16 * hi];
    o0 = wmma16(pa, v0, o0);  // d = 0..15
    o1 = wmma16(pa, v1, o1);  // d = 16..31

    __syncthreads();  // next chunk's stage complete; this buf safe to overwrite
  }

  // final row-sum reduction and normalization
#pragma unroll
  for (int mask = 1; mask < 16; mask <<= 1)
#pragma unroll
    for (int r = 0; r < 8; ++r) l[r] += __shfl_xor(l[r], mask, 32);
#pragma unroll
  for (int r = 0; r < 8; ++r) {
    const float inv = 1.0f / l[r];
    const int row = 8 * hi + r;
    att[qbase + (size_t)row * DHEAD + lo]      = (_Float16)(o0[r] * inv);
    att[qbase + (size_t)row * DHEAD + 16 + lo] = (_Float16)(o1[r] * inv);
  }
}

// ---------------------------------------------------------------------------
// Kernel 5: output projection (M=256, K=128, N=4096 per batch) + bias, f32 out.
// K-steps align with heads since att is stored [b,h,s,d]. Waves of a block
// share (b, s-group) so the att B operand is reused 8x from cache.
// ---------------------------------------------------------------------------
__global__ void __launch_bounds__(256) out_gemm(const _Float16* __restrict__ att,
                                                const _Float16* __restrict__ wo,
                                                const float* __restrict__ bias,
                                                float* __restrict__ out) {
  const int lane = threadIdx.x & 31;
  const int lo = lane & 15, hi = lane >> 4;
  const int warp = threadIdx.x >> 5;
  const int mg = blockIdx.x & 1;           // 2 m-groups of 8 tiles
  const int ng = (blockIdx.x >> 1) & 63;   // 64 s-groups
  const int b  = blockIdx.x >> 7;
  const int mt = mg * 8 + warp;            // 16 tiles over M=256
  const int o0 = mt * 16;
  const int s0 = ng * 64;

  v8f acc[4] = {};
#pragma unroll
  for (int ks = 0; ks < 4; ++ks) {  // ks == head
    const int c0 = ks * 32;
    v16h a = loadA(wo + (size_t)(o0 + lo) * HID + c0 + 16 * hi);
#pragma unroll
    for (int sub = 0; sub < 4; ++sub) {
      const int s = s0 + sub * 16 + lo;
      v16h bv = *(const v16h*)(att + (((size_t)b * HEADS + ks) * SEQ + s) * DHEAD +
                               16 * hi);
      acc[sub] = wmma16(a, bv, acc[sub]);
    }
  }
  float bi[8];
#pragma unroll
  for (int r = 0; r < 8; ++r) bi[r] = bias[o0 + 8 * hi + r];
#pragma unroll
  for (int sub = 0; sub < 4; ++sub)
#pragma unroll
    for (int r = 0; r < 8; ++r) {
      const int o = o0 + 8 * hi + r;
      const int s = s0 + sub * 16 + lo;
      out[((size_t)b * CDIM + o) * SEQ + s] = acc[sub][r] + bi[r];
    }
}

// ---------------------------------------------------------------------------
extern "C" void kernel_launch(void* const* d_in, const int* in_sizes, int n_in,
                              void* d_out, int out_size, void* d_ws, size_t ws_size,
                              hipStream_t stream) {
  (void)in_sizes; (void)n_in; (void)out_size; (void)ws_size;
  const float* x    = (const float*)d_in[0];  // [4,256,64,64]
  const float* wqkv = (const float*)d_in[1];  // [384,256]
  const float* wout = (const float*)d_in[2];  // [256,128]
  const float* bout = (const float*)d_in[3];  // [256]
  float* out = (float*)d_out;                 // [4,256,64,64]

  // workspace carve (f16), all offsets 64B-aligned; total ~24.3 MB
  _Float16* xt  = (_Float16*)d_ws;                          // [b,s,c]
  _Float16* wq  = xt + (size_t)BATCH * SEQ * CDIM;          // [384,256]
  _Float16* wo  = wq + 384 * 256;                           // [256,128]
  _Float16* qb  = wo + 256 * 128;                           // [b,h,s,d] (scaled)
  _Float16* kb  = qb + (size_t)BATCH * HEADS * SEQ * DHEAD; // [b,h,s,d]
  _Float16* vt  = kb + (size_t)BATCH * HEADS * SEQ * DHEAD; // [b,h,d,s]
  _Float16* att = vt + (size_t)BATCH * HEADS * SEQ * DHEAD; // [b,h,s,d]

  conv_weights<<<dim3(384), dim3(256), 0, stream>>>(wqkv, wout, wq, wo);
  xpose<<<dim3(SEQ / 32, CDIM / 32, BATCH), dim3(32, 32), 0, stream>>>(x, xt);
  qkv_gemm<<<dim3(768), dim3(256), 0, stream>>>(xt, wq, qb, kb, vt);
  flash_attn<<<dim3(512), dim3(256), 0, stream>>>(qb, kb, vt, att);
  out_gemm<<<dim3(512), dim3(256), 0, stream>>>(att, wo, bout, out);
}
